// LoRAExplicitGRUCell_58995670778339
// MI455X (gfx1250) — matrix operand order repaired and checked by reading, not compiled, not run on hardware
//
#include <hip/hip_runtime.h>

typedef __attribute__((ext_vector_type(2)))  float        v2f;
typedef __attribute__((ext_vector_type(4)))  float        v4f;
typedef __attribute__((ext_vector_type(8)))  float        v8f;
typedef __attribute__((ext_vector_type(4)))  int          v4i;
typedef __attribute__((ext_vector_type(4)))  unsigned int v4u;
typedef __attribute__((ext_vector_type(8)))  __bf16       v8bf;
typedef __attribute__((ext_vector_type(16))) __bf16       v16bf;

#define B_ 2048
#define D_ 1024
#define F_ 1024
#define S_ 64
#define R_ 16

#define NMAT ((size_t)D_ * F_)       // elements per weight matrix
#define ACT  ((size_t)B_ * D_)       // elements per activation matrix

#if defined(__has_builtin)
#  if __has_builtin(__builtin_amdgcn_global_load_async_to_lds_b128)
#    define HAS_ASYNC_LDS 1
#  endif
#endif
#ifndef HAS_ASYNC_LDS
#  define HAS_ASYNC_LDS 0
#endif

#define AS1 __attribute__((address_space(1)))
#define AS3 __attribute__((address_space(3)))

__device__ __forceinline__ void wait_async_lds() {
#if HAS_ASYNC_LDS
#  if __has_builtin(__builtin_amdgcn_s_wait_asynccnt)
    __builtin_amdgcn_s_wait_asynccnt(0);
#  else
    asm volatile("s_wait_asynccnt 0" ::: "memory");
#  endif
#endif
}

// ---------------------------------------------------------------------------
// Kernel 1: low[g][b][r] = sum_d src[b,d] * la_g[ids[b], d, r]
// One wave32 per (g, b) pair.  lane -> (r = lane&15, parity = lane>>4).
// ---------------------------------------------------------------------------
__global__ __launch_bounds__(256) void lora_low_kernel(
    const float* __restrict__ x, const float* __restrict__ h,
    const int* __restrict__ ids,
    const float* __restrict__ la0, const float* __restrict__ la1,
    const float* __restrict__ la2, const float* __restrict__ la3,
    const float* __restrict__ la4, const float* __restrict__ la5,
    float* __restrict__ low)
{
    const int wave = (int)((blockIdx.x * blockDim.x + threadIdx.x) >> 5);
    const int lane = (int)(threadIdx.x & 31);
    const int g = wave / B_;
    const int b = wave % B_;
    const int r = lane & 15;
    const int half = lane >> 4;

    const float* src = (g < 3) ? x : h;
    const float* la;
    switch (g) {
        case 0: la = la0; break;
        case 1: la = la1; break;
        case 2: la = la2; break;
        case 3: la = la3; break;
        case 4: la = la4; break;
        default: la = la5; break;
    }
    const int id = ids[b];
    const float* arow = la + (size_t)id * (D_ * R_);
    const float* srow = src + (size_t)b * D_;

    float acc = 0.f;
#pragma unroll 8
    for (int d = half; d < D_; d += 2) {
        acc = fmaf(srow[d], arow[d * R_ + r], acc);
    }
    acc += __shfl_xor(acc, 16);
    if (half == 0) low[((size_t)g * B_ + b) * R_ + r] = acc;
}

// ---------------------------------------------------------------------------
// Kernel W: transpose + hi/lo bf16 split of the six DxF fp32 weights.
//   wt layout: for g in 0..5:  [2g] = wT_hi (FxD bf16), [2g+1] = wT_lo.
// ---------------------------------------------------------------------------
__global__ __launch_bounds__(256) void weight_split_kernel(
    const float* __restrict__ w0, const float* __restrict__ w1,
    const float* __restrict__ w2, const float* __restrict__ w3,
    const float* __restrict__ w4, const float* __restrict__ w5,
    __bf16* __restrict__ wt)
{
    __shared__ float tile[64][65];
    const int g = (int)blockIdx.y;
    const float* w;
    switch (g) {
        case 0: w = w0; break;
        case 1: w = w1; break;
        case 2: w = w2; break;
        case 3: w = w3; break;
        case 4: w = w4; break;
        default: w = w5; break;
    }
    __bf16* hi = wt + (size_t)(2 * g) * NMAT;
    __bf16* lo = hi + NMAT;

    const int f0 = (int)(blockIdx.x & 15) * 64;
    const int k0 = (int)(blockIdx.x >> 4) * 64;
    const int tx = (int)(threadIdx.x & 63);
    const int ty = (int)(threadIdx.x >> 6);

#pragma unroll
    for (int i = 0; i < 16; ++i) {
        const int kk = ty + 4 * i;
        tile[kk][tx] = w[(size_t)(k0 + kk) * F_ + (size_t)(f0 + tx)];
    }
    __syncthreads();
#pragma unroll
    for (int i = 0; i < 16; ++i) {
        const int ff = ty + 4 * i;
        const float v = tile[tx][ff];
        const __bf16 hv = (__bf16)v;
        const __bf16 lv = (__bf16)(v - (float)hv);
        const size_t o = (size_t)(f0 + ff) * D_ + (size_t)(k0 + tx);
        hi[o] = hv;
        lo[o] = lv;
    }
}

// ---------------------------------------------------------------------------
// Kernel A: split x and h into bf16 hi/lo (row-major BxD).
//   act layout: [0]=x_hi [1]=x_lo [2]=h_hi [3]=h_lo, each ACT elements.
// ---------------------------------------------------------------------------
__global__ __launch_bounds__(256) void act_split_kernel(
    const float* __restrict__ x, const float* __restrict__ h,
    __bf16* __restrict__ act)
{
    const size_t tid = (size_t)blockIdx.x * blockDim.x + threadIdx.x;
    const size_t e0 = tid * 8;
    const bool isH = e0 >= ACT;
    const float* src = isH ? h : x;
    const size_t off = isH ? (e0 - ACT) : e0;
    __bf16* hi = act + (isH ? 2 * ACT : (size_t)0) + off;
    __bf16* lo = act + (isH ? 3 * ACT : ACT) + off;

    float t[8];
    *(v4f*)(t + 0) = *(const v4f*)(src + off);
    *(v4f*)(t + 4) = *(const v4f*)(src + off + 4);
    v8bf vh, vl;
#pragma unroll
    for (int e = 0; e < 8; ++e) {
        const __bf16 hv = (__bf16)t[e];
        vh[e] = hv;
        vl[e] = (__bf16)(t[e] - (float)hv);
    }
    *(v8bf*)hi = vh;
    *(v8bf*)lo = vl;
}

// ---------------------------------------------------------------------------
// Shared fused epilogue: bias + rank-16 LoRA delta + GRU pointwise combine.
// ---------------------------------------------------------------------------
__device__ __forceinline__ float sigmoid_(float v) {
    return 1.0f / (1.0f + __expf(-v));
}
__device__ __forceinline__ float tanh_(float v) {
    v = fminf(fmaxf(v, -15.0f), 15.0f);
    float e = __expf(2.0f * v);
    return (e - 1.0f) / (e + 1.0f);
}

__device__ __forceinline__ void gru_epilogue(
    int b0, int f0, int n, int khalf,
    const v8f& acc0, const v8f& acc1, const v8f& acc2,
    const v8f& acc3, const v8f& acc4, const v8f& acc5,
    const int* __restrict__ ids, const float* __restrict__ low,
    const float* __restrict__ bias_ir, const float* __restrict__ bias_iu,
    const float* __restrict__ bias_ic,
    const float* __restrict__ lb_ir, const float* __restrict__ lb_iu,
    const float* __restrict__ lb_ic, const float* __restrict__ lb_rr,
    const float* __restrict__ lb_ru, const float* __restrict__ lb_rc,
    const float* __restrict__ h, float* __restrict__ out)
{
    const int f = f0 + n;
    const float b_ir = bias_ir[f];
    const float b_iu = bias_iu[f];
    const float b_ic = bias_ic[f];

#pragma unroll
    for (int i = 0; i < 8; ++i) {
        const int bb = b0 + i + 8 * khalf;
        const int id = ids[bb];
        const float* l0 = low + ((size_t)0 * B_ + bb) * R_;
        const float* l1 = low + ((size_t)1 * B_ + bb) * R_;
        const float* l2 = low + ((size_t)2 * B_ + bb) * R_;
        const float* l3 = low + ((size_t)3 * B_ + bb) * R_;
        const float* l4 = low + ((size_t)4 * B_ + bb) * R_;
        const float* l5 = low + ((size_t)5 * B_ + bb) * R_;
        const size_t lbo = ((size_t)id * R_) * F_ + (size_t)f;

        float d0 = 0.f, d1 = 0.f, d2 = 0.f, d3 = 0.f, d4 = 0.f, d5 = 0.f;
#pragma unroll
        for (int r = 0; r < R_; ++r) {
            const size_t o = lbo + (size_t)r * F_;
            d0 = fmaf(l0[r], lb_ir[o], d0);
            d1 = fmaf(l1[r], lb_iu[o], d1);
            d2 = fmaf(l2[r], lb_ic[o], d2);
            d3 = fmaf(l3[r], lb_rr[o], d3);
            d4 = fmaf(l4[r], lb_ru[o], d4);
            d5 = fmaf(l5[r], lb_rc[o], d5);
        }

        const float g_ir = acc0[i] + b_ir + d0;
        const float g_iu = acc1[i] + b_iu + d1;
        const float g_ic = acc2[i] + b_ic + d2;
        const float g_rr = acc3[i] + d3;
        const float g_ru = acc4[i] + d4;
        const float g_rc = acc5[i] + d5;

        const float rg = sigmoid_(g_ir + g_rr);
        const float ug = sigmoid_(g_iu + g_ru);
        const float cg = tanh_(g_ic + rg * g_rc);
        const float hc = h[(size_t)bb * F_ + f];
        out[(size_t)bb * F_ + f] = (1.0f - ug) * cg + ug * hc;
    }
}

// ---------------------------------------------------------------------------
// WMMA helpers.
// ---------------------------------------------------------------------------
#define WMMA_BF16(ACC, A, Bv)                                                  \
    ACC = __builtin_amdgcn_wmma_f32_16x16x32_bf16(false, (A), false, (Bv),     \
                                                  (short)0, (ACC), false, false)

__device__ __forceinline__ v16bf ld16bf(const __bf16* p) {
    v8bf a = *(const v8bf*)p;
    v8bf b = *(const v8bf*)(p + 8);
    return __builtin_shufflevector(a, b, 0, 1, 2, 3, 4, 5, 6, 7,
                                         8, 9, 10, 11, 12, 13, 14, 15);
}

__device__ __forceinline__ v16bf frag2(const unsigned char* smem,
                                       unsigned o0, unsigned o1) {
    v8bf a = *(const v8bf*)(smem + o0);
    v8bf b = *(const v8bf*)(smem + o1);
    return __builtin_shufflevector(a, b, 0, 1, 2, 3, 4, 5, 6, 7,
                                         8, 9, 10, 11, 12, 13, 14, 15);
}

// ---------------------------------------------------------------------------
// Kernel 2 (primary): LDS-staged, double-buffered, async-copy 3xBF16 GEMM.
// Block = 512 threads = 16 waves covering a 64x64 macro-tile (4x4 of 16x16).
// Per K-chunk of 32: stage 16 arrays (4 act + 12 split weights), 64 rows x
// 64 B each, into LDS (80 B row stride -> conflict-free ds_load_b128), then
// 18 x V_WMMA_F32_16X16X32_BF16 per wave.
// ---------------------------------------------------------------------------
#define ROWSTRIDE 80u
#define ARRBYTES  (64u * ROWSTRIDE)      // 5120 B per array
#define BUFBYTES  (16u * ARRBYTES)       // 81920 B per buffer
#define LDSBYTES  (2u * BUFBYTES)        // 163840 B total

__device__ __forceinline__ void stage_tile(
    unsigned char* smem, int buf, int k0, int tid,
    const __bf16* __restrict__ act, const __bf16* __restrict__ wt,
    int blockB, int blockF)
{
    const unsigned base = (unsigned)buf * BUFBYTES;
#pragma unroll
    for (int j = 0; j < 8; ++j) {
        const int c = tid + 512 * j;      // 4096 chunks of 16 B
        const int a = c >> 8;             // array 0..15
        const int rem = c & 255;
        const int r = rem >> 2;           // row 0..63
        const int q = rem & 3;            // 16-B piece in row
        const __bf16* src;
        int row;
        if (a < 4) { src = act + (size_t)a * ACT;       row = blockB + r; }
        else       { src = wt + (size_t)(a - 4) * NMAT; row = blockF + r; }
        const __bf16* g = src + (size_t)row * D_ + (size_t)(k0 + 8 * q);
        const unsigned l = base + (unsigned)a * ARRBYTES
                         + (unsigned)r * ROWSTRIDE + (unsigned)q * 16u;
#if HAS_ASYNC_LDS
        __builtin_amdgcn_global_load_async_to_lds_b128(
            (AS1 v4i*)(void*)g, (AS3 v4i*)(void*)(smem + l), 0, 0);
#else
        *(v4u*)(smem + l) = *(const v4u*)g;
#endif
    }
}

__global__ __launch_bounds__(512) void gru_wmma_bf16_lds_kernel(
    const __bf16* __restrict__ act, const __bf16* __restrict__ wt,
    const int* __restrict__ ids,
    const float* __restrict__ bias_ir, const float* __restrict__ bias_iu,
    const float* __restrict__ bias_ic,
    const float* __restrict__ lb_ir, const float* __restrict__ lb_iu,
    const float* __restrict__ lb_ic, const float* __restrict__ lb_rr,
    const float* __restrict__ lb_ru, const float* __restrict__ lb_rc,
    const float* __restrict__ low,
    const float* __restrict__ h, float* __restrict__ out)
{
    extern __shared__ unsigned char smem[];

    const int tid = (int)threadIdx.x;
    const int wave = tid >> 5;
    const int lane = tid & 31;
    const int mb = wave & 3;              // b sub-tile
    const int nb = wave >> 2;             // f sub-tile
    const int fgrp = (int)blockIdx.x & 15;
    const int bgrp = (int)blockIdx.x >> 4;
    const int blockB = bgrp * 64;
    const int blockF = fgrp * 64;
    const int n = lane & 15;
    const int khalf = lane >> 4;

    v8f acc0 = {}, acc1 = {}, acc2 = {}, acc3 = {}, acc4 = {}, acc5 = {};

    const unsigned mr = (unsigned)(16 * mb + n) * ROWSTRIDE;   // A row base
    const unsigned fr = (unsigned)(16 * nb + n) * ROWSTRIDE;   // B row base
    const unsigned aoff = (unsigned)(32 * khalf);              // A run offsets
    const unsigned boff = (unsigned)(32 * khalf);              // B run offset

    stage_tile(smem, 0, 0, tid, act, wt, blockB, blockF);
    wait_async_lds();
    __syncthreads();

    for (int it = 0; it < D_ / 32; ++it) {
        const int cur = it & 1;
        if (it + 1 < D_ / 32)
            stage_tile(smem, cur ^ 1, (it + 1) * 32, tid, act, wt, blockB, blockF);

        const unsigned base = (unsigned)cur * BUFBYTES;
        const unsigned axb = base + 0u * ARRBYTES + mr;
        const unsigned alb = base + 1u * ARRBYTES + mr;
        const unsigned hxb = base + 2u * ARRBYTES + mr;
        const unsigned hlb = base + 3u * ARRBYTES + mr;
        const v16bf axh = frag2(smem, axb + aoff, axb + 16u + aoff);
        const v16bf axl = frag2(smem, alb + aoff, alb + 16u + aoff);
        const v16bf ahh = frag2(smem, hxb + aoff, hxb + 16u + aoff);
        const v16bf ahl = frag2(smem, hlb + aoff, hlb + 16u + aoff);

        const unsigned b0b = base + 4u * ARRBYTES + fr + boff;
        const v16bf b0h = frag2(smem, b0b + 0u * ARRBYTES, b0b + 0u * ARRBYTES + 16u);
        const v16bf b0l = frag2(smem, b0b + 1u * ARRBYTES, b0b + 1u * ARRBYTES + 16u);
        const v16bf b1h = frag2(smem, b0b + 2u * ARRBYTES, b0b + 2u * ARRBYTES + 16u);
        const v16bf b1l = frag2(smem, b0b + 3u * ARRBYTES, b0b + 3u * ARRBYTES + 16u);
        const v16bf b2h = frag2(smem, b0b + 4u * ARRBYTES, b0b + 4u * ARRBYTES + 16u);
        const v16bf b2l = frag2(smem, b0b + 5u * ARRBYTES, b0b + 5u * ARRBYTES + 16u);
        const v16bf b3h = frag2(smem, b0b + 6u * ARRBYTES, b0b + 6u * ARRBYTES + 16u);
        const v16bf b3l = frag2(smem, b0b + 7u * ARRBYTES, b0b + 7u * ARRBYTES + 16u);
        const v16bf b4h = frag2(smem, b0b + 8u * ARRBYTES, b0b + 8u * ARRBYTES + 16u);
        const v16bf b4l = frag2(smem, b0b + 9u * ARRBYTES, b0b + 9u * ARRBYTES + 16u);
        const v16bf b5h = frag2(smem, b0b + 10u * ARRBYTES, b0b + 10u * ARRBYTES + 16u);
        const v16bf b5l = frag2(smem, b0b + 11u * ARRBYTES, b0b + 11u * ARRBYTES + 16u);

        // hi*hi terms (per-accumulator reuse distance 6)
        WMMA_BF16(acc0, axh, b0h);
        WMMA_BF16(acc1, axh, b1h);
        WMMA_BF16(acc2, axh, b2h);
        WMMA_BF16(acc3, ahh, b3h);
        WMMA_BF16(acc4, ahh, b4h);
        WMMA_BF16(acc5, ahh, b5h);
        // hi*lo terms
        WMMA_BF16(acc0, axh, b0l);
        WMMA_BF16(acc1, axh, b1l);
        WMMA_BF16(acc2, axh, b2l);
        WMMA_BF16(acc3, ahh, b3l);
        WMMA_BF16(acc4, ahh, b4l);
        WMMA_BF16(acc5, ahh, b5l);
        // lo*hi terms
        WMMA_BF16(acc0, axl, b0h);
        WMMA_BF16(acc1, axl, b1h);
        WMMA_BF16(acc2, axl, b2h);
        WMMA_BF16(acc3, ahl, b3h);
        WMMA_BF16(acc4, ahl, b4h);
        WMMA_BF16(acc5, ahl, b5h);

        wait_async_lds();
        __syncthreads();
    }

    gru_epilogue(blockB + 16 * mb, blockF + 16 * nb, n, khalf,
                 acc0, acc1, acc2, acc3, acc4, acc5,
                 ids, low, bias_ir, bias_iu, bias_ic,
                 lb_ir, lb_iu, lb_ic, lb_rr, lb_ru, lb_rc, h, out);
}

// ---------------------------------------------------------------------------
// Kernel 2 (mid fallback): per-wave bf16 WMMA, A split on the fly.
// ---------------------------------------------------------------------------
__device__ __forceinline__ void split_a(const float* p0, const float* p1,
                                        v16bf& hi, v16bf& lo) {
    float t[16];
    *(v4f*)(t + 0)  = *(const v4f*)(p0);
    *(v4f*)(t + 4)  = *(const v4f*)(p0 + 4);
    *(v4f*)(t + 8)  = *(const v4f*)(p1);
    *(v4f*)(t + 12) = *(const v4f*)(p1 + 4);
#pragma unroll
    for (int e = 0; e < 16; ++e) {
        const __bf16 hv = (__bf16)t[e];
        hi[e] = hv;
        lo[e] = (__bf16)(t[e] - (float)hv);
    }
}

__global__ __launch_bounds__(256) void gru_wmma_bf16_kernel(
    const float* __restrict__ x, const float* __restrict__ h,
    const int* __restrict__ ids,
    const __bf16* __restrict__ wt,
    const float* __restrict__ bias_ir, const float* __restrict__ bias_iu,
    const float* __restrict__ bias_ic,
    const float* __restrict__ lb_ir, const float* __restrict__ lb_iu,
    const float* __restrict__ lb_ic, const float* __restrict__ lb_rr,
    const float* __restrict__ lb_ru, const float* __restrict__ lb_rc,
    const float* __restrict__ low, float* __restrict__ out)
{
    const int wave = (int)(threadIdx.x >> 5);
    const int lane = (int)(threadIdx.x & 31);
    const int tile = (int)blockIdx.x * 8 + wave;
    const int tF = tile & ((F_ / 16) - 1);
    const int tB = tile >> 6;
    const int f0 = tF * 16;
    const int b0 = tB * 16;
    const int n = lane & 15;
    const int khalf = lane >> 4;

    v8f acc0 = {}, acc1 = {}, acc2 = {}, acc3 = {}, acc4 = {}, acc5 = {};

    const float* xrow = x + (size_t)(b0 + n) * D_;
    const float* hrow = h + (size_t)(b0 + n) * F_;

    const size_t wro = (size_t)(f0 + n) * D_ + (size_t)(16 * khalf);
    const __bf16* w0h = wt + 0 * NMAT + wro;  const __bf16* w0l = wt + 1 * NMAT + wro;
    const __bf16* w1h = wt + 2 * NMAT + wro;  const __bf16* w1l = wt + 3 * NMAT + wro;
    const __bf16* w2h = wt + 4 * NMAT + wro;  const __bf16* w2l = wt + 5 * NMAT + wro;
    const __bf16* w3h = wt + 6 * NMAT + wro;  const __bf16* w3l = wt + 7 * NMAT + wro;
    const __bf16* w4h = wt + 8 * NMAT + wro;  const __bf16* w4l = wt + 9 * NMAT + wro;
    const __bf16* w5h = wt + 10 * NMAT + wro; const __bf16* w5l = wt + 11 * NMAT + wro;

    for (int k0 = 0; k0 < D_; k0 += 32) {
        const int ka = k0 + 16 * khalf;
        const int kb = k0 + 16 * khalf + 8;

        v16bf axh, axl, ahh, ahl;
        split_a(xrow + ka, xrow + kb, axh, axl);
        split_a(hrow + ka, hrow + kb, ahh, ahl);

        const v16bf b0h = ld16bf(w0h + k0), b0l = ld16bf(w0l + k0);
        const v16bf b1h = ld16bf(w1h + k0), b1l = ld16bf(w1l + k0);
        const v16bf b2h = ld16bf(w2h + k0), b2l = ld16bf(w2l + k0);
        const v16bf b3h = ld16bf(w3h + k0), b3l = ld16bf(w3l + k0);
        const v16bf b4h = ld16bf(w4h + k0), b4l = ld16bf(w4l + k0);
        const v16bf b5h = ld16bf(w5h + k0), b5l = ld16bf(w5l + k0);

        WMMA_BF16(acc0, axh, b0h);
        WMMA_BF16(acc1, axh, b1h);
        WMMA_BF16(acc2, axh, b2h);
        WMMA_BF16(acc3, ahh, b3h);
        WMMA_BF16(acc4, ahh, b4h);
        WMMA_BF16(acc5, ahh, b5h);
        WMMA_BF16(acc0, axh, b0l);
        WMMA_BF16(acc1, axh, b1l);
        WMMA_BF16(acc2, axh, b2l);
        WMMA_BF16(acc3, ahh, b3l);
        WMMA_BF16(acc4, ahh, b4l);
        WMMA_BF16(acc5, ahh, b5l);
        WMMA_BF16(acc0, axl, b0h);
        WMMA_BF16(acc1, axl, b1h);
        WMMA_BF16(acc2, axl, b2h);
        WMMA_BF16(acc3, ahl, b3h);
        WMMA_BF16(acc4, ahl, b4h);
        WMMA_BF16(acc5, ahl, b5h);
    }

    gru_epilogue(b0, f0, n, khalf, acc0, acc1, acc2, acc3, acc4, acc5,
                 ids, low, bias_ir, bias_iu, bias_ic,
                 lb_ir, lb_iu, lb_ic, lb_rr, lb_ru, lb_rc, h, out);
}

// ---------------------------------------------------------------------------
// Kernel 2 (small-workspace fallback): fp32 V_WMMA_F32_16X16X4_F32 path.
// ---------------------------------------------------------------------------
__global__ __launch_bounds__(256) void gru_wmma_f32_kernel(
    const float* __restrict__ x, const float* __restrict__ h,
    const int* __restrict__ ids,
    const float* __restrict__ w_ir, const float* __restrict__ bias_ir, const float* __restrict__ lb_ir,
    const float* __restrict__ w_iu, const float* __restrict__ bias_iu, const float* __restrict__ lb_iu,
    const float* __restrict__ w_ic, const float* __restrict__ bias_ic, const float* __restrict__ lb_ic,
    const float* __restrict__ w_rr, const float* __restrict__ lb_rr,
    const float* __restrict__ w_ru, const float* __restrict__ lb_ru,
    const float* __restrict__ w_rc, const float* __restrict__ lb_rc,
    const float* __restrict__ low, float* __restrict__ out)
{
    const int wave = (int)(threadIdx.x >> 5);
    const int lane = (int)(threadIdx.x & 31);
    const int tile = (int)blockIdx.x * 8 + wave;
    const int tF = tile & ((F_ / 16) - 1);
    const int tB = tile >> 6;
    const int f0 = tF * 16;
    const int b0 = tB * 16;
    const int n = lane & 15;
    const int khalf = lane >> 4;

    v8f acc0 = {}, acc1 = {}, acc2 = {}, acc3 = {}, acc4 = {}, acc5 = {};

    const float* xrow = x + (size_t)(b0 + n) * D_;
    const float* hrow = h + (size_t)(b0 + n) * F_;

#pragma unroll 2
    for (int k0 = 0; k0 < D_; k0 += 4) {
        const int ka = k0 + 2 * khalf;
        v2f ax = *(const v2f*)(xrow + ka);
        v2f ah = *(const v2f*)(hrow + ka);

        const size_t o0 = (size_t)ka * F_ + (size_t)(f0 + n);
        const size_t o1 = o0 + F_;
        v2f bir = { w_ir[o0], w_ir[o1] };
        v2f biu = { w_iu[o0], w_iu[o1] };
        v2f bic = { w_ic[o0], w_ic[o1] };
        v2f brr = { w_rr[o0], w_rr[o1] };
        v2f bru = { w_ru[o0], w_ru[o1] };
        v2f brc = { w_rc[o0], w_rc[o1] };

        acc0 = __builtin_amdgcn_wmma_f32_16x16x4_f32(false, ax, false, bir, (short)0, acc0, false, false);
        acc1 = __builtin_amdgcn_wmma_f32_16x16x4_f32(false, ax, false, biu, (short)0, acc1, false, false);
        acc2 = __builtin_amdgcn_wmma_f32_16x16x4_f32(false, ax, false, bic, (short)0, acc2, false, false);
        acc3 = __builtin_amdgcn_wmma_f32_16x16x4_f32(false, ah, false, brr, (short)0, acc3, false, false);
        acc4 = __builtin_amdgcn_wmma_f32_16x16x4_f32(false, ah, false, bru, (short)0, acc4, false, false);
        acc5 = __builtin_amdgcn_wmma_f32_16x16x4_f32(false, ah, false, brc, (short)0, acc5, false, false);
    }

    gru_epilogue(b0, f0, n, khalf, acc0, acc1, acc2, acc3, acc4, acc5,
                 ids, low, bias_ir, bias_iu, bias_ic,
                 lb_ir, lb_iu, lb_ic, lb_rr, lb_ru, lb_rc, h, out);
}

extern "C" void kernel_launch(void* const* d_in, const int* in_sizes, int n_in,
                              void* d_out, int out_size, void* d_ws, size_t ws_size,
                              hipStream_t stream) {
    (void)in_sizes; (void)n_in; (void)out_size;

    const float* carry   = (const float*)d_in[0];
    const float* inputs  = (const float*)d_in[1];
    const int*   ids     = (const int*)d_in[2];
    const float* w_ir    = (const float*)d_in[3];
    const float* bias_ir = (const float*)d_in[4];
    const float* la_ir   = (const float*)d_in[5];
    const float* lb_ir   = (const float*)d_in[6];
    const float* w_iu    = (const float*)d_in[7];
    const float* bias_iu = (const float*)d_in[8];
    const float* la_iu   = (const float*)d_in[9];
    const float* lb_iu   = (const float*)d_in[10];
    const float* w_ic    = (const float*)d_in[11];
    const float* bias_ic = (const float*)d_in[12];
    const float* la_ic   = (const float*)d_in[13];
    const float* lb_ic   = (const float*)d_in[14];
    const float* w_rr    = (const float*)d_in[15];
    const float* la_rr   = (const float*)d_in[16];
    const float* lb_rr   = (const float*)d_in[17];
    const float* w_ru    = (const float*)d_in[18];
    const float* la_ru   = (const float*)d_in[19];
    const float* lb_ru   = (const float*)d_in[20];
    const float* w_rc    = (const float*)d_in[21];
    const float* la_rc   = (const float*)d_in[22];
    const float* lb_rc   = (const float*)d_in[23];

    const size_t LOW_BYTES = (size_t)6 * B_ * R_ * sizeof(float);   // 768 KiB
    const size_t WT_BYTES  = 12 * NMAT * 2;                         // 24 MiB
    const size_t ACT_BYTES = 4 * ACT * 2;                           // 16 MiB
    float*  low = (float*)d_ws;
    __bf16* wt  = (__bf16*)((char*)d_ws + LOW_BYTES);
    __bf16* act = (__bf16*)((char*)d_ws + LOW_BYTES + WT_BYTES);
    float*  out = (float*)d_out;

    lora_low_kernel<<<dim3((6 * B_) / 8), dim3(256), 0, stream>>>(
        inputs, carry, ids, la_ir, la_iu, la_ic, la_rr, la_ru, la_rc, low);

    if (ws_size >= LOW_BYTES + WT_BYTES + ACT_BYTES) {
        weight_split_kernel<<<dim3(256, 6), dim3(256), 0, stream>>>(
            w_ir, w_iu, w_ic, w_rr, w_ru, w_rc, wt);
        act_split_kernel<<<dim3((unsigned)(2 * ACT / (256 * 8))), dim3(256), 0, stream>>>(
            inputs, carry, act);
        gru_wmma_bf16_lds_kernel<<<dim3(512), dim3(512), LDSBYTES, stream>>>(
            act, wt, ids, bias_ir, bias_iu, bias_ic,
            lb_ir, lb_iu, lb_ic, lb_rr, lb_ru, lb_rc,
            low, carry, out);
    } else if (ws_size >= LOW_BYTES + WT_BYTES) {
        weight_split_kernel<<<dim3(256, 6), dim3(256), 0, stream>>>(
            w_ir, w_iu, w_ic, w_rr, w_ru, w_rc, wt);
        gru_wmma_bf16_kernel<<<dim3(((B_ / 16) * (F_ / 16)) / 8), dim3(256), 0, stream>>>(
            inputs, carry, ids, wt,
            bias_ir, bias_iu, bias_ic,
            lb_ir, lb_iu, lb_ic, lb_rr, lb_ru, lb_rc,
            low, out);
    } else {
        gru_wmma_f32_kernel<<<dim3(((B_ / 16) * (F_ / 16)) / 8), dim3(256), 0, stream>>>(
            inputs, carry, ids,
            w_ir, bias_ir, lb_ir,
            w_iu, bias_iu, lb_iu,
            w_ic, bias_ic, lb_ic,
            w_rr, lb_rr,
            w_ru, lb_ru,
            w_rc, lb_rc,
            low, out);
    }
}
